// GAT_84825604096157
// MI455X (gfx1250) — matrix operand order repaired and checked by reading, not compiled, not run on hardware
//
#include <hip/hip_runtime.h>
#include <hip/hip_bf16.h>

#define NN 100000
#define EE 1600000
#define ET (EE + NN)          // edges + self loops
#define FD 128                // feature dim at every layer boundary (H*C = 128)

typedef __attribute__((ext_vector_type(16))) __bf16 v16bf;
typedef __attribute__((ext_vector_type(8)))  float  v8f;

// ---------------------------------------------------------------------------
// Y[nrows x 128] = X[nrows x 128] @ W[128 x 128], f32 in/out.
// bf16 hi/lo split: X = Xhi + Xlo, W = Whi + Wlo,
//   Y ~= Xhi*Whi + Xhi*Wlo + Xlo*Whi    (error ~2^-16 relative)
//
// Block = 256 threads = 8 waves; block owns a 128-row x 128-col tile;
// wave w owns rows [w*16, w*16+16) x all 128 cols (8 accum tiles of 16x16).
//
// Weights are PRE-PACKED in LDS in the exact wave32 B-fragment layout
// ([kb][nt][lane] -> 16 contiguous bf16), so the hot loop reads each
// fragment with 2x ds_load_b128 instead of 16x ds_load_u16.
// ---------------------------------------------------------------------------
__global__ __launch_bounds__(256)
void gemm128_bf16x3_kernel(const float* __restrict__ X,
                           const float* __restrict__ W,
                           float* __restrict__ Y, int nrows)
{
    // packed B fragments: 4 kb-blocks x 8 n-tiles x 32 lanes x 16 bf16 = 32KB each
    __shared__ v16bf Bhi[4][8][32];
    __shared__ v16bf Blo[4][8][32];

    const int tid = threadIdx.x;

    // ---- pack phase: 1024 fragments, 4 per thread ----
    for (int f = tid; f < 1024; f += 256) {
        int lane = f & 31;
        int nt   = (f >> 5) & 7;
        int kb   = f >> 8;
        int half = lane >> 4;
        int ncol = nt * 16 + (lane & 15);
        v16bf hi, lo;
#pragma unroll
        for (int v = 0; v < 8; ++v) {
            int k = kb * 32 + 2 * v + 16 * half;   // B layout: half splits K 0-15 / 16-31
#pragma unroll
            for (int j = 0; j < 2; ++j) {
                float w  = W[(size_t)(k + j) * FD + ncol];
                __bf16 h = (__bf16)w;
                hi[2 * v + j] = h;
                lo[2 * v + j] = (__bf16)(w - (float)h);
            }
        }
        Bhi[kb][nt][lane] = hi;
        Blo[kb][nt][lane] = lo;
    }
    __syncthreads();

    const int wave = tid >> 5;
    const int lane = tid & 31;
    const int half = lane >> 4;      // 0: lanes 0-15, 1: lanes 16-31
    const int l16  = lane & 15;
    const int rowStrip = blockIdx.x * 128 + wave * 16;

    v8f acc[8] = {};                 // 8 n-tiles of 16x16 f32

    for (int kb = 0; kb < 4; ++kb) { // K = 128 in chunks of 32
        // ---- A fragment: 16x32 bf16, contiguous k order (same as B packing).
        // For this half of the wave: element e -> xrow[e] (k = 16*half + e)
        int arow = rowStrip + l16;
        if (arow >= nrows) arow = nrows - 1;      // clamp (stores are guarded)
        const float4* xp = (const float4*)(X + (size_t)arow * FD + kb * 32 + 16 * half);
        float4 q0 = xp[0], q1 = xp[1];            // xrow[0..7]
        float4 q2 = xp[2], q3 = xp[3];            // xrow[8..15]
        float af[16] = { q0.x, q0.y, q0.z, q0.w, q1.x, q1.y, q1.z, q1.w,
                         q2.x, q2.y, q2.z, q2.w, q3.x, q3.y, q3.z, q3.w };
        v16bf ahi, alo;
#pragma unroll
        for (int e = 0; e < 16; ++e) {
            __bf16 h = (__bf16)af[e];
            ahi[e] = h;
            alo[e] = (__bf16)(af[e] - (float)h);
        }
#pragma unroll
        for (int nt = 0; nt < 8; ++nt) {
            v16bf bhi = Bhi[kb][nt][lane];        // 2x ds_load_b128
            v16bf blo = Blo[kb][nt][lane];        // 2x ds_load_b128
            acc[nt] = __builtin_amdgcn_wmma_f32_16x16x32_bf16(
                false, ahi, false, bhi, (short)0, acc[nt], false, false);
            acc[nt] = __builtin_amdgcn_wmma_f32_16x16x32_bf16(
                false, ahi, false, blo, (short)0, acc[nt], false, false);
            acc[nt] = __builtin_amdgcn_wmma_f32_16x16x32_bf16(
                false, alo, false, bhi, (short)0, acc[nt], false, false);
        }
    }

    // D layout: lane -> col = l16, VGPR v -> row = v + 8*half
#pragma unroll
    for (int nt = 0; nt < 8; ++nt) {
#pragma unroll
        for (int v = 0; v < 8; ++v) {
            int row = rowStrip + v + 8 * half;
            if (row < nrows)
                Y[(size_t)row * FD + nt * 16 + l16] = acc[nt][v];
        }
    }
}

// ---------------------------------------------------------------------------
// float atomic max via monotonic int/uint trick (works with -inf init)
// ---------------------------------------------------------------------------
__device__ __forceinline__ void atomicMaxF32(float* addr, float val) {
    if (val >= 0.0f)
        atomicMax((int*)addr, __float_as_int(val));
    else
        atomicMin((unsigned int*)addr, __float_as_uint(val));
}

// init: agg = 0 (N*128), m = -inf, den = 0 (N*4)
__global__ void init_kernel(float* agg, float* m, float* den) {
    int i = blockIdx.x * blockDim.x + threadIdx.x;
    if (i < NN * FD) agg[i] = 0.0f;
    if (i < NN * 4) { m[i] = -__builtin_inff(); den[i] = 0.0f; }
}

// ---------------------------------------------------------------------------
// One wave per edge. lane = channel (0..31), 4 heads per lane.
// logits[e][h] = sum_c att[h][c] * leakyrelu(xl[src][h][c] + xr[dst][h][c])
// plus atomic segment-max into m[dst][h].
// ---------------------------------------------------------------------------
__global__ __launch_bounds__(256)
void edge_logits_kernel(const float* __restrict__ XLp, const float* __restrict__ XRp,
                        const int* __restrict__ es, const int* __restrict__ ed,
                        const float* __restrict__ att,
                        float* __restrict__ lg, float* __restrict__ mbuf)
{
    int wid = blockIdx.x * 8 + (threadIdx.x >> 5);
    if (wid >= ET) return;
    int lane = threadIdx.x & 31;
    int s, d;
    if (wid < EE) { s = es[wid]; d = ed[wid]; } else { s = wid - EE; d = s; }
    const float* xl = XLp + (size_t)s * FD;
    const float* xr = XRp + (size_t)d * FD;

    float logit[4];
#pragma unroll
    for (int h = 0; h < 4; ++h) {
        float v = xl[h * 32 + lane] + xr[h * 32 + lane];
        v = (v > 0.0f) ? v : 0.2f * v;              // LeakyReLU, slope 0.2
        float p = v * att[h * 32 + lane];
#pragma unroll
        for (int off = 16; off > 0; off >>= 1)
            p += __shfl_xor(p, off, 32);
        logit[h] = p;
    }
    if (lane == 0) {
#pragma unroll
        for (int h = 0; h < 4; ++h) {
            lg[(size_t)wid * 4 + h] = logit[h];
            atomicMaxF32(&mbuf[(size_t)d * 4 + h], logit[h]);
        }
    }
}

// ex = exp(logit - m[dst]);  den[dst] += ex;  (lg overwritten with ex)
__global__ void edge_softmax_kernel(const int* __restrict__ ed,
                                    const float* __restrict__ mbuf,
                                    float* __restrict__ lg,
                                    float* __restrict__ den)
{
    long long t = (long long)blockIdx.x * blockDim.x + threadIdx.x;
    if (t >= (long long)ET * 4) return;
    int e = (int)(t >> 2), h = (int)(t & 3);
    int d = (e < EE) ? ed[e] : (e - EE);
    float ex = __expf(lg[t] - mbuf[(size_t)d * 4 + h]);
    lg[t] = ex;
    atomicAdd(&den[(size_t)d * 4 + h], ex);
}

// agg[dst] += (ex/den[dst]) * xl[src]  -- one wave per edge, lane = channel
__global__ __launch_bounds__(256)
void edge_scatter_kernel(const float* __restrict__ XLp,
                         const int* __restrict__ es, const int* __restrict__ ed,
                         const float* __restrict__ lg, const float* __restrict__ den,
                         float* __restrict__ agg)
{
    int wid = blockIdx.x * 8 + (threadIdx.x >> 5);
    if (wid >= ET) return;
    int lane = threadIdx.x & 31;
    int s, d;
    if (wid < EE) { s = es[wid]; d = ed[wid]; } else { s = wid - EE; d = s; }

    float alpha[4];
#pragma unroll
    for (int h = 0; h < 4; ++h)
        alpha[h] = lg[(size_t)wid * 4 + h] / den[(size_t)d * 4 + h];

    const float* xl = XLp + (size_t)s * FD;
    float* ao = agg + (size_t)d * FD;
#pragma unroll
    for (int h = 0; h < 4; ++h)
        atomicAdd(&ao[h * 32 + lane], alpha[h] * xl[h * 32 + lane]);
}

// layers 1,2: h = relu(agg + b), in place
__global__ void bias_relu_kernel(float* __restrict__ h, const float* __restrict__ b) {
    int i = blockIdx.x * blockDim.x + threadIdx.x;
    if (i < NN * FD) {
        float v = h[i] + b[i & 127];
        h[i] = v > 0.0f ? v : 0.0f;
    }
}

// layer 3: out[n][c] = relu(mean_h(agg[n][h][c]) + b3[c])
__global__ void head_mean_kernel(const float* __restrict__ agg,
                                 const float* __restrict__ b3,
                                 float* __restrict__ out) {
    int i = blockIdx.x * blockDim.x + threadIdx.x;
    if (i < NN * 32) {
        int n = i >> 5, c = i & 31;
        const float* a = agg + (size_t)n * FD;
        float v = 0.25f * (a[c] + a[32 + c] + a[64 + c] + a[96 + c]) + b3[c];
        out[i] = v > 0.0f ? v : 0.0f;
    }
}

// ---------------------------------------------------------------------------
extern "C" void kernel_launch(void* const* d_in, const int* in_sizes, int n_in,
                              void* d_out, int out_size, void* d_ws, size_t ws_size,
                              hipStream_t stream) {
    (void)in_sizes; (void)n_in; (void)out_size; (void)ws_size;

    const float* x    = (const float*)d_in[0];
    const int*   ei   = (const int*)d_in[1];
    const int*   esrc = ei;
    const int*   edst = ei + EE;
    const float* Wl[3]  = { (const float*)d_in[2],  (const float*)d_in[6],  (const float*)d_in[10] };
    const float* Wr[3]  = { (const float*)d_in[3],  (const float*)d_in[7],  (const float*)d_in[11] };
    const float* att[3] = { (const float*)d_in[4],  (const float*)d_in[8],  (const float*)d_in[12] };
    const float* bia[3] = { (const float*)d_in[5],  (const float*)d_in[9],  (const float*)d_in[13] };
    float* out = (float*)d_out;

    // workspace layout (f32)
    float* XL  = (float*)d_ws;                 // N*128
    float* XR  = XL  + (size_t)NN * FD;        // N*128
    float* AGG = XR  + (size_t)NN * FD;        // N*128: layer1 h, reused as layer3 agg
    float* H2  = AGG + (size_t)NN * FD;        // N*128: layer2 h
    float* LG  = H2  + (size_t)NN * FD;        // ET*4: logits -> ex
    float* M   = LG  + (size_t)ET * 4;         // N*4
    float* DEN = M   + (size_t)NN * 4;         // N*4

    const int gemmBlocks = (NN + 127) / 128;           // 782
    const int initBlocks = (NN * FD + 255) / 256;      // 50000
    const int edgeBlocks = (ET + 7) / 8;               // 212500 (8 waves/block)
    const long long smTh  = (long long)ET * 4;
    const int smBlocks   = (int)((smTh + 255) / 256);

    const float* hin[3]  = { x, AGG, H2 };
    float*       hagg[3] = { AGG, H2, AGG };

    for (int L = 0; L < 3; ++L) {
        gemm128_bf16x3_kernel<<<gemmBlocks, 256, 0, stream>>>(hin[L], Wl[L], XL, NN);
        gemm128_bf16x3_kernel<<<gemmBlocks, 256, 0, stream>>>(hin[L], Wr[L], XR, NN);
        init_kernel<<<initBlocks, 256, 0, stream>>>(hagg[L], M, DEN);
        edge_logits_kernel<<<edgeBlocks, 256, 0, stream>>>(XL, XR, esrc, edst, att[L], LG, M);
        edge_softmax_kernel<<<smBlocks, 256, 0, stream>>>(edst, M, LG, DEN);
        edge_scatter_kernel<<<edgeBlocks, 256, 0, stream>>>(XL, esrc, edst, LG, DEN, hagg[L]);
        if (L < 2)
            bias_relu_kernel<<<initBlocks, 256, 0, stream>>>(hagg[L], bia[L]);
        else
            head_mean_kernel<<<(NN * 32 + 255) / 256, 256, 0, stream>>>(hagg[L], bia[L], out);
    }
}
